// AttentionSpikingNetwork_26671746908706
// MI455X (gfx1250) — matrix operand order repaired and checked by reading, not compiled, not run on hardware
//
#include <hip/hip_runtime.h>
#include <hip/hip_bf16.h>

// ---------------------------------------------------------------------------
// Spiking attention network, CDNA5 (gfx1250) WMMA implementation.
//   embed -> LIF -> {Q,K,V} -> softmax(QK^T/8)V -> +spk1 -> FFN -> LIF -> head
// GEMMs run on v_wmma_f32_16x16x32_f16 (f32 accumulate). LDS staging uses the
// CDNA5 async direct-to-LDS path (global_load_async_to_lds_b128 + ASYNCcnt),
// double-buffered in the GEMM K-loop so DMA overlaps WMMA compute.
// Spikes {0,1} are exact in f16. Padded dims: 784->800, 600->640, 200->256.
// ---------------------------------------------------------------------------

typedef __attribute__((ext_vector_type(16))) _Float16 v16h;
typedef __attribute__((ext_vector_type(8)))  float    v8f;

#define NB    64
#define SEQ   512
#define DIN   784
#define DEMB  600
#define DQK   64
#define DH2   200
#define DOUT  10

#define PK_IN  800           // padded D_IN  (K of embed GEMM, %32==0)
#define PN_EMB 640           // padded D_EMB (%64==0)
#define PN_H2  256           // padded D_H2  (%64==0)
#define NTOK   (NB * SEQ)    // 32768

static __device__ __forceinline__ v8f vzero8() {
  v8f z = {0.f, 0.f, 0.f, 0.f, 0.f, 0.f, 0.f, 0.f};
  return z;
}

// CDNA5 async global->LDS copy (16B per lane). VDST VGPR carries the LDS byte
// address; ASYNCcnt tracks completion (CDNA5 ISA).
static __device__ __forceinline__ void async_copy_b128(void* lds_dst,
                                                       const void* gsrc) {
  unsigned lds_off = (unsigned)(unsigned long long)lds_dst;
  asm volatile("global_load_async_to_lds_b128 %0, %1, off"
               :: "v"(lds_off), "v"(gsrc)
               : "memory");
}
static __device__ __forceinline__ void wait_async0() {
  asm volatile("s_wait_asynccnt 0x0" ::: "memory");
}

// ---------------------------------------------------------------------------
// Setup kernels: max-reduce (for the conditional /255), f32->f16 packers.
// ---------------------------------------------------------------------------
__global__ void init_max_kernel(unsigned* maxbits) { *maxbits = 0u; }

__global__ __launch_bounds__(256) void reduce_max_kernel(
    const float* __restrict__ x, long n, unsigned* __restrict__ maxbits) {
  long stride = (long)gridDim.x * 256;
  float v = 0.f;
  for (long i = (long)blockIdx.x * 256 + threadIdx.x; i < n; i += stride)
    v = fmaxf(v, x[i]);
  #pragma unroll
  for (int off = 16; off >= 1; off >>= 1) v = fmaxf(v, __shfl_xor(v, off, 32));
  if ((threadIdx.x & 31) == 0) atomicMax(maxbits, __float_as_uint(v));  // x>=0
}

__global__ __launch_bounds__(256) void convert_x_kernel(
    const float* __restrict__ x, _Float16* __restrict__ xh,
    const unsigned* __restrict__ maxbits) {
  float scale = (__uint_as_float(*maxbits) > 1.f) ? (1.f / 255.f) : 1.f;
  long idx = (long)blockIdx.x * 256 + threadIdx.x;
  long tot = (long)NTOK * PK_IN;
  if (idx >= tot) return;
  long r = idx / PK_IN;
  int  c = (int)(idx % PK_IN);
  float v = (c < DIN) ? x[r * DIN + c] * scale : 0.f;
  xh[idx] = (_Float16)v;
}

__global__ __launch_bounds__(256) void convert_w_kernel(
    const float* __restrict__ W, _Float16* __restrict__ Wh,
    int realN, int realK, int padK, long tot) {
  long idx = (long)blockIdx.x * 256 + threadIdx.x;
  if (idx >= tot) return;
  int n = (int)(idx / padK);
  int k = (int)(idx % padK);
  float v = (n < realN && k < realK) ? W[(long)n * realK + k] : 0.f;
  Wh[idx] = (_Float16)v;
}

// ---------------------------------------------------------------------------
// Tiled WMMA GEMM:  out[M,Npad] = act( A[M,K] * W[Npad,K]^T + bias )
// Block tile 128x64, BK=32, 256 threads (8 waves). Wave w owns output rows
// [w*16, w*16+16) x 64 cols (4 accumulators). A and W staged in LDS by the
// async direct-to-LDS path, double-buffered: tile t+1's DMA is issued before
// tile t's WMMAs so the copy overlaps compute; s_wait_asynccnt 0 + barrier
// publishes the next buffer.  mode: 0 = bias only, 1 = LIF spike (v > thr).
// Fragment layouts follow ISA 7.12.2 (16-bit A 16x32, B 32x16, f32 C/D).
// ---------------------------------------------------------------------------
__global__ __launch_bounds__(256) void gemm_wmma_kernel(
    const _Float16* __restrict__ A, int lda,
    const _Float16* __restrict__ W, int ldw,
    const float* __restrict__ bias, int realN,
    _Float16* __restrict__ out, int ldo,
    int K, int mode, float thr) {
  __shared__ alignas(16) _Float16 As[2][128][32];   // 16 KB
  __shared__ alignas(16) _Float16 Ws[2][64][32];    //  8 KB

  const int tid  = threadIdx.x;
  const int wave = tid >> 5;
  const int lane = tid & 31;
  const int m16  = lane & 15;
  const int half = lane >> 4;
  const long m0  = (long)blockIdx.y * 128;
  const int  n0  = blockIdx.x * 64;

  // Per-thread staging coordinates (3 async b128 copies per thread per tile).
  const int ar0 = tid >> 2,           akc0 = (tid & 3) * 8;          // A chunk 0
  const int ar1 = (tid + 256) >> 2,   akc1 = (tid & 3) * 8;          // A chunk 1
  const int wr  = tid >> 2,           wkc  = (tid & 3) * 8;          // W chunk

  v8f acc[4];
  #pragma unroll
  for (int i = 0; i < 4; ++i) acc[i] = vzero8();

  auto stage = [&](int buf, int k0) {
    async_copy_b128(&As[buf][ar0][akc0], &A[(m0 + ar0) * (long)lda + k0 + akc0]);
    async_copy_b128(&As[buf][ar1][akc1], &A[(m0 + ar1) * (long)lda + k0 + akc1]);
    async_copy_b128(&Ws[buf][wr][wkc],   &W[(n0 + wr) * (long)ldw + k0 + wkc]);
  };

  const int nk = K >> 5;
  stage(0, 0);
  wait_async0();
  __syncthreads();

  for (int t = 0; t < nk; ++t) {
    const int buf = t & 1;
    if (t + 1 < nk) stage(buf ^ 1, (t + 1) * 32);   // overlap DMA with WMMA

    // A fragment (rows wave*16 .. +15).
    v16h afrag;
    #pragma unroll
    for (int j = 0; j < 8; ++j) {
      int k = 16 * (j >> 2) + 8 * half + 2 * (j & 3);
      afrag[2 * j]     = As[buf][wave * 16 + m16][k];
      afrag[2 * j + 1] = As[buf][wave * 16 + m16][k + 1];
    }
    #pragma unroll
    for (int ns = 0; ns < 4; ++ns) {
      v16h bfrag;
      #pragma unroll
      for (int j = 0; j < 8; ++j) {
        int k = 16 * (j >> 2) + 8 * half + 2 * (j & 3);   // B[k][n] = W[n][k]
        bfrag[2 * j]     = Ws[buf][ns * 16 + m16][k];
        bfrag[2 * j + 1] = Ws[buf][ns * 16 + m16][k + 1];
      }
      acc[ns] = __builtin_amdgcn_wmma_f32_16x16x32_f16(
          false, afrag, false, bfrag, (short)0, acc[ns], false, false);
    }

    wait_async0();      // next tile's LDS data landed (per-wave)
    __syncthreads();    // publish to all waves; buf^1 safe to read next iter
  }

  // Epilogue: D layout -> lane n = lane%16, VGPR r -> row r + 8*(lane/16).
  #pragma unroll
  for (int ns = 0; ns < 4; ++ns) {
    #pragma unroll
    for (int r = 0; r < 8; ++r) {
      int m   = r + 8 * half;
      int col = n0 + ns * 16 + m16;
      float v = acc[ns][r];
      if (col < realN) v += bias[col];
      if (mode == 1) v = (v > thr) ? 1.f : 0.f;
      out[(m0 + wave * 16 + m) * (long)ldo + col] = (_Float16)v;
    }
  }
}

// ---------------------------------------------------------------------------
// Fused attention:  out = softmax(Q K^T / 8) V + spk1   (per batch)
// One block = one (batch, 16-query tile). 256 threads / 8 waves.
//   Phase A: scores 16x512 via WMMA into LDS (wave w owns keys [64w,64w+64))
//   Phase B: softmax (16 lanes per row, shfl reductions), P in f16 LDS
//   Phase C: P @ V with the 32x640 V k-slab async-DMA'd into LDS (unioned
//            with the dead score buffer); epilogue scales by 1/rowsum, +spk1.
// All of V/K per batch (~0.7 MB f16) lives in L2 (192 MB) across the 32
// query-tiles of a batch, so the re-reads here are L2 traffic, not HBM.
// ---------------------------------------------------------------------------
__global__ __launch_bounds__(256) void attn_wmma_kernel(
    const _Float16* __restrict__ Qh, const _Float16* __restrict__ Kh,
    const _Float16* __restrict__ Vh, const _Float16* __restrict__ spk1,
    _Float16* __restrict__ outp) {
  __shared__ alignas(16) union {
    float    Sc[16][SEQ];        // scores (phases A/B)      32 KB
    _Float16 Vt[32][PN_EMB];     // V k-slab (phase C)       40 KB
  } sh;
  __shared__ alignas(16) _Float16 P[16][SEQ];  // softmax numerators, 16 KB
  __shared__ float rstat[16];                  // 1 / rowsum

  const int b   = blockIdx.y;
  const int q0  = blockIdx.x * 16;
  const int tid = threadIdx.x;
  const int wave = tid >> 5;
  const int lane = tid & 31;
  const int m16  = lane & 15;
  const int half = lane >> 4;
  const long qbase = (long)b * SEQ + q0;
  const long kbase = (long)b * SEQ;

  // ---- Phase A: scores = Q K^T * (1/8) -----------------------------------
  {
    v8f acc[4];
    #pragma unroll
    for (int i = 0; i < 4; ++i) acc[i] = vzero8();
    #pragma unroll
    for (int k0 = 0; k0 < DQK; k0 += 32) {
      v16h afrag;
      #pragma unroll
      for (int j = 0; j < 8; ++j) {
        int k = 16 * (j >> 2) + 8 * half + 2 * (j & 3);
        afrag[2 * j]     = Qh[(qbase + m16) * DQK + k0 + k];
        afrag[2 * j + 1] = Qh[(qbase + m16) * DQK + k0 + k + 1];
      }
      #pragma unroll
      for (int ns = 0; ns < 4; ++ns) {
        int n0 = wave * 64 + ns * 16;
        v16h bfrag;
        #pragma unroll
        for (int j = 0; j < 8; ++j) {
          int k = 16 * (j >> 2) + 8 * half + 2 * (j & 3);   // B[k][n] = K[n][k]
          bfrag[2 * j]     = Kh[(kbase + n0 + m16) * DQK + k0 + k];
          bfrag[2 * j + 1] = Kh[(kbase + n0 + m16) * DQK + k0 + k + 1];
        }
        acc[ns] = __builtin_amdgcn_wmma_f32_16x16x32_f16(
            false, afrag, false, bfrag, (short)0, acc[ns], false, false);
      }
    }
    const float scale = 0.125f;   // 1/sqrt(64)
    #pragma unroll
    for (int ns = 0; ns < 4; ++ns)
      #pragma unroll
      for (int r = 0; r < 8; ++r) {
        int m = r + 8 * half;
        sh.Sc[m][wave * 64 + ns * 16 + m16] = acc[ns][r] * scale;
      }
  }
  __syncthreads();

  // ---- Phase B: row softmax (16 lanes per row) ---------------------------
  {
    int row = tid >> 4, l16 = tid & 15;
    float mx = -3.0e38f;
    for (int c = l16; c < SEQ; c += 16) mx = fmaxf(mx, sh.Sc[row][c]);
    #pragma unroll
    for (int off = 8; off >= 1; off >>= 1) mx = fmaxf(mx, __shfl_xor(mx, off, 16));
    float sum = 0.f;
    for (int c = l16; c < SEQ; c += 16) {
      float e = __expf(sh.Sc[row][c] - mx);
      P[row][c] = (_Float16)e;
      sum += e;
    }
    #pragma unroll
    for (int off = 8; off >= 1; off >>= 1) sum += __shfl_xor(sum, off, 16);
    if (l16 == 0) rstat[row] = 1.f / sum;
  }
  __syncthreads();

  // ---- Phase C: P @ V (V slab async-DMA'd into LDS), + spk1 --------------
  v8f acc[5];   // wave w owns N-subtiles w, w+8, ..., w+32  (5 * 8 = 40)
  #pragma unroll
  for (int i = 0; i < 5; ++i) acc[i] = vzero8();

  for (int k0 = 0; k0 < SEQ; k0 += 32) {
    // Stage V rows [k0, k0+32) x 640 : 2560 x 16B async copies, 10/thread.
    #pragma unroll
    for (int i = 0; i < 10; ++i) {
      int c  = tid + 256 * i;
      int r  = c / 80;
      int kc = (c % 80) * 8;
      async_copy_b128(&sh.Vt[r][kc],
                      &Vh[(kbase + k0 + r) * (long)PN_EMB + kc]);
    }
    wait_async0();
    __syncthreads();

    v16h afrag;
    #pragma unroll
    for (int j = 0; j < 8; ++j) {
      int k = 16 * (j >> 2) + 8 * half + 2 * (j & 3);
      afrag[2 * j]     = P[m16][k0 + k];
      afrag[2 * j + 1] = P[m16][k0 + k + 1];
    }
    #pragma unroll
    for (int i = 0; i < 5; ++i) {
      int n0 = (wave + 8 * i) * 16;
      v16h bfrag;
      #pragma unroll
      for (int j = 0; j < 8; ++j) {
        int k = 16 * (j >> 2) + 8 * half + 2 * (j & 3);
        bfrag[2 * j]     = sh.Vt[k][n0 + m16];
        bfrag[2 * j + 1] = sh.Vt[k + 1][n0 + m16];
      }
      acc[i] = __builtin_amdgcn_wmma_f32_16x16x32_f16(
          false, afrag, false, bfrag, (short)0, acc[i], false, false);
    }
    __syncthreads();
  }

  #pragma unroll
  for (int i = 0; i < 5; ++i) {
    int n0 = (wave + 8 * i) * 16;
    #pragma unroll
    for (int r = 0; r < 8; ++r) {
      int m = r + 8 * half;
      long row = qbase + m;
      float v = acc[i][r] * rstat[m] +
                (float)spk1[row * (long)PN_EMB + n0 + m16];
      outp[row * (long)PN_EMB + n0 + m16] = (_Float16)v;
    }
  }
}

// ---------------------------------------------------------------------------
// Head: cur3 = spk2 @ W3^T + b3 ; spk3 = (cur3>0.3) ; mem3 = cur3 - spk3*0.3
// 10-wide output -> plain VALU kernel, one thread per token.
// ---------------------------------------------------------------------------
__global__ __launch_bounds__(256) void head_kernel(
    const _Float16* __restrict__ spk2, const float* __restrict__ W3,
    const float* __restrict__ b3, float* __restrict__ out) {
  long t = (long)blockIdx.x * 256 + threadIdx.x;
  if (t >= NTOK) return;
  const _Float16* row = spk2 + t * PN_H2;
  #pragma unroll
  for (int o = 0; o < DOUT; ++o) {
    float acc = b3[o];
    const float* w = W3 + o * DH2;
    for (int k = 0; k < DH2; ++k) acc += (float)row[k] * w[k];
    float spk = (acc > 0.3f) ? 1.f : 0.f;
    out[t * DOUT + o] = spk;                                  // spk3
    out[(long)NTOK * DOUT + t * DOUT + o] = acc - spk * 0.3f; // mem3
  }
}

// ---------------------------------------------------------------------------
// Launcher
// ---------------------------------------------------------------------------
extern "C" void kernel_launch(void* const* d_in, const int* in_sizes, int n_in,
                              void* d_out, int out_size, void* d_ws, size_t ws_size,
                              hipStream_t stream) {
  const float* x  = (const float*)d_in[0];
  const float* We = (const float*)d_in[1];
  const float* be = (const float*)d_in[2];
  const float* Wq = (const float*)d_in[3];
  const float* bq = (const float*)d_in[4];
  const float* Wk = (const float*)d_in[5];
  const float* bk = (const float*)d_in[6];
  const float* Wv = (const float*)d_in[7];
  const float* bv = (const float*)d_in[8];
  const float* W2 = (const float*)d_in[9];
  const float* b2 = (const float*)d_in[10];
  const float* W3 = (const float*)d_in[11];
  const float* b3 = (const float*)d_in[12];
  float* out = (float*)d_out;

  char* ws = (char*)d_ws;
  size_t off = 0;
  auto alloc = [&](size_t bytes) -> char* {
    char* p = ws + off;
    off += (bytes + 255) & ~(size_t)255;
    return p;
  };
  unsigned* maxbits = (unsigned*)alloc(sizeof(unsigned));
  _Float16* xh   = (_Float16*)alloc((size_t)NTOK * PK_IN * 2);
  _Float16* Weh  = (_Float16*)alloc((size_t)PN_EMB * PK_IN * 2);
  _Float16* Wqh  = (_Float16*)alloc((size_t)DQK * PN_EMB * 2);
  _Float16* Wkh  = (_Float16*)alloc((size_t)DQK * PN_EMB * 2);
  _Float16* Wvh  = (_Float16*)alloc((size_t)PN_EMB * PN_EMB * 2);
  _Float16* W2h  = (_Float16*)alloc((size_t)PN_H2 * PN_EMB * 2);
  _Float16* spk1 = (_Float16*)alloc((size_t)NTOK * PN_EMB * 2);
  _Float16* Qh   = (_Float16*)alloc((size_t)NTOK * DQK * 2);
  _Float16* Kh   = (_Float16*)alloc((size_t)NTOK * DQK * 2);
  _Float16* Vh   = (_Float16*)alloc((size_t)NTOK * PN_EMB * 2);
  _Float16* s2in = (_Float16*)alloc((size_t)NTOK * PN_EMB * 2);
  _Float16* spk2 = (_Float16*)alloc((size_t)NTOK * PN_H2 * 2);

  // 0) input max (for the conditional /255), then pack x and weights to f16.
  init_max_kernel<<<1, 1, 0, stream>>>(maxbits);
  reduce_max_kernel<<<1024, 256, 0, stream>>>(x, (long)NTOK * DIN, maxbits);
  convert_x_kernel<<<((long)NTOK * PK_IN + 255) / 256, 256, 0, stream>>>(x, xh, maxbits);

  auto convw = [&](const float* src, _Float16* dst, int rN, int rK, int pN, int pK) {
    long tot = (long)pN * pK;
    convert_w_kernel<<<(tot + 255) / 256, 256, 0, stream>>>(src, dst, rN, rK, pK, tot);
  };
  convw(We, Weh, DEMB, DIN,  PN_EMB, PK_IN);
  convw(Wq, Wqh, DQK,  DEMB, DQK,    PN_EMB);
  convw(Wk, Wkh, DQK,  DEMB, DQK,    PN_EMB);
  convw(Wv, Wvh, DEMB, DEMB, PN_EMB, PN_EMB);
  convw(W2, W2h, DH2,  DEMB, PN_H2,  PN_EMB);

  dim3 blk(256);
  // 1) embed + LIF(0.5)  -> spk1 [NTOK, 640]
  gemm_wmma_kernel<<<dim3(PN_EMB / 64, NTOK / 128), blk, 0, stream>>>(
      xh, PK_IN, Weh, PK_IN, be, DEMB, spk1, PN_EMB, PK_IN, 1, 0.5f);
  // 2) Q, K [NTOK, 64] ; V [NTOK, 640]
  gemm_wmma_kernel<<<dim3(DQK / 64, NTOK / 128), blk, 0, stream>>>(
      spk1, PN_EMB, Wqh, PN_EMB, bq, DQK, Qh, DQK, PN_EMB, 0, 0.f);
  gemm_wmma_kernel<<<dim3(DQK / 64, NTOK / 128), blk, 0, stream>>>(
      spk1, PN_EMB, Wkh, PN_EMB, bk, DQK, Kh, DQK, PN_EMB, 0, 0.f);
  gemm_wmma_kernel<<<dim3(PN_EMB / 64, NTOK / 128), blk, 0, stream>>>(
      spk1, PN_EMB, Wvh, PN_EMB, bv, DEMB, Vh, PN_EMB, PN_EMB, 0, 0.f);
  // 3) fused attention + residual -> s2in [NTOK, 640]
  attn_wmma_kernel<<<dim3(SEQ / 16, NB), blk, 0, stream>>>(Qh, Kh, Vh, spk1, s2in);
  // 4) FFN + LIF(0.3) -> spk2 [NTOK, 256]
  gemm_wmma_kernel<<<dim3(PN_H2 / 64, NTOK / 128), blk, 0, stream>>>(
      s2in, PN_EMB, W2h, PN_EMB, b2, DH2, spk2, PN_H2, PN_EMB, 1, 0.3f);
  // 5) head -> (spk3, mem3) f32 into d_out
  head_kernel<<<dim3(NTOK / 256), blk, 0, stream>>>(spk2, W3, b3, out);
}
